// SVDLlamaBlock_85529978733236
// MI455X (gfx1250) — compile-verified, matrix operand order checked by reading, not run on hardware
//
#include <hip/hip_runtime.h>
#include <hip/hip_bf16.h>

#define B_    2
#define T_    1024
#define D_    4096
#define H_    32
#define HK_   8
#define DH_   128
#define REP_  4
#define INTER_ 14336
#define RQ_   64
#define RKV_  64
#define RO_   512
#define RFF_  1024
#define MTOK_ (B_*T_)      // 2048
#define EPS_  1e-5f

typedef __attribute__((ext_vector_type(16))) __bf16 v16bf;
typedef __attribute__((ext_vector_type(8)))  __bf16 v8bf;
typedef __attribute__((ext_vector_type(8)))  float  v8f;

// ---------- bf16 helpers (manual RNE, no reliance on cast lowering) ----------
__device__ __forceinline__ __bf16 f2bf(float f) {
  union { float f; unsigned u; } x; x.f = f;
  unsigned r = x.u + 0x7FFFu + ((x.u >> 16) & 1u);
  union { unsigned short s; __bf16 b; } y; y.s = (unsigned short)(r >> 16);
  return y.b;
}
__device__ __forceinline__ float bf2f(__bf16 b) {
  union { unsigned short s; __bf16 b; } x; x.b = b;
  union { unsigned u; float f; } y; y.u = ((unsigned)x.s) << 16;
  return y.f;
}

// Load one 16-element bf16 fragment: lo 8 at base, hi 8 at base+16 elements.
// With base = row + kchunk + (lane>=16 ? 8 : 0), this matches the CDNA5
// 16-bit A/B fragment layout (lanes 0-15: K 0-7 & 16-23; lanes 16-31: 8-15 & 24-31).
__device__ __forceinline__ v16bf load_frag(const __bf16* base) {
  union { v16bf v; struct { v8bf lo, hi; } s; } u;
  u.s.lo = *(const v8bf*)(base);
  u.s.hi = *(const v8bf*)(base + 16);
  return u.v;
}

// ---------------------------- fp32 -> bf16 cast ----------------------------
__global__ void cast_f32_bf16(const float* __restrict__ in, __bf16* __restrict__ out, long long n) {
  long long i = (long long)blockIdx.x * blockDim.x + threadIdx.x;
  long long stride = (long long)gridDim.x * blockDim.x;
  for (; i < n; i += stride) out[i] = f2bf(in[i]);
}

// ------------------------------- RMSNorm -----------------------------------
__global__ __launch_bounds__(256) void rmsnorm_kernel(const float* __restrict__ x,
                                                      const float* __restrict__ w,
                                                      __bf16* __restrict__ out, int Dn) {
  int row = blockIdx.x;
  const float* xp = x + (size_t)row * Dn;
  float s = 0.f;
  for (int i = threadIdx.x; i < Dn; i += 256) { float v = xp[i]; s += v * v; }
  #pragma unroll
  for (int m = 16; m; m >>= 1) s += __shfl_xor(s, m, 32);
  __shared__ float red[8];
  if ((threadIdx.x & 31) == 0) red[threadIdx.x >> 5] = s;
  __syncthreads();
  float tot = 0.f;
  #pragma unroll
  for (int i = 0; i < 8; ++i) tot += red[i];
  float rs = rsqrtf(tot / (float)Dn + EPS_);
  __bf16* op = out + (size_t)row * Dn;
  for (int i = threadIdx.x; i < Dn; i += 256) op[i] = f2bf(xp[i] * rs * w[i]);
}

// -------------------------------- RoPE -------------------------------------
// x layout: (B*T, heads, Dh) bf16; rotate pairs (f, f+64) per head/token.
__global__ void rope_kernel(__bf16* __restrict__ x, const float* __restrict__ inv_freq,
                            int heads, long long total) {
  long long i = (long long)blockIdx.x * blockDim.x + threadIdx.x;
  if (i >= total) return;
  int f = (int)(i & 63);
  long long rest = i >> 6;
  int hh = (int)(rest % heads);
  long long m = rest / heads;
  int t = (int)(m % T_);
  float ang = (float)t * inv_freq[f];
  float c, sn; __sincosf(ang, &sn, &c);
  __bf16* p = x + ((size_t)m * heads + hh) * DH_;
  float x1 = bf2f(p[f]), x2 = bf2f(p[f + 64]);
  p[f]      = f2bf(x1 * c - x2 * sn);
  p[f + 64] = f2bf(x2 * c + x1 * sn);
}

// --------------------------- WMMA GEMM: C = A @ W^T -------------------------
// A: (M,K) bf16 row-major, W: (N,K) bf16 row-major. 8 waves/block. Each wave
// computes NT 16x16 tiles along N, reusing its A fragment NT times (A-frag
// reuse lifts compute density to ~NT wmma per (2+2*NT) b128 loads).
// Block covers 16 rows x (128*NT) cols. Batched via blockIdx.z strides.
// OUTF32=1: C fp32, epilogue adds residual R (same layout).
template <int NT, int OUTF32>
__global__ __launch_bounds__(256) void gemm_wmma_bf16(
    const __bf16* __restrict__ A, const __bf16* __restrict__ W,
    void* __restrict__ Cout, const float* __restrict__ R,
    int K, int lda, int ldw, int ldc,
    long long sA, long long sW, long long sC, long long sR) {
  const int wave = threadIdx.x >> 5;
  const int lane = threadIdx.x & 31;
  const int hf   = lane >> 4;
  const int l16  = lane & 15;
  const int ntile0 = (blockIdx.x * 8 + wave) * NT;
  const int mtile  = blockIdx.y;
  const long long z = blockIdx.z;

  const __bf16* Ap = A + z * sA + (size_t)(mtile * 16 + l16) * lda + hf * 8;
  const __bf16* Wp = W + z * sW + (size_t)(ntile0 * 16 + l16) * ldw + hf * 8;

  v8f c[NT];
  #pragma unroll
  for (int nt = 0; nt < NT; ++nt) c[nt] = (v8f){0.f,0.f,0.f,0.f,0.f,0.f,0.f,0.f};

  for (int kb = 0; kb < K; kb += 32) {
    v16bf a = load_frag(Ap + kb);
    #pragma unroll
    for (int nt = 0; nt < NT; ++nt) {
      v16bf b = load_frag(Wp + (size_t)nt * 16 * ldw + kb);
      c[nt] = __builtin_amdgcn_wmma_f32_16x16x32_bf16(false, a, false, b, (short)0, c[nt], false, false);
    }
  }

  const int row0 = mtile * 16 + hf * 8;
  #pragma unroll
  for (int nt = 0; nt < NT; ++nt) {
    const int col = (ntile0 + nt) * 16 + l16;
    if (OUTF32) {
      float* C = (float*)Cout + z * sC;
      const float* Rp = R + z * sR;
      #pragma unroll
      for (int j = 0; j < 8; ++j) {
        size_t idx = (size_t)(row0 + j) * ldc + col;
        C[idx] = c[nt][j] + Rp[idx];
      }
    } else {
      __bf16* C = (__bf16*)Cout + z * sC;
      #pragma unroll
      for (int j = 0; j < 8; ++j)
        C[(size_t)(row0 + j) * ldc + col] = f2bf(c[nt][j]);
    }
  }
}

// ----------------------- Flash attention (causal GQA) -----------------------
// Q: (B*T, H, Dh) bf16; K,V: (B*T, Hk, Dh) bf16; O: (B*T, H, Dh) bf16.
// Block = 8 waves; wave w owns query rows [qblock+16w, +16). Keys stream in
// 32-wide tiles; V tile staged transposed in LDS for PV B-fragments; P tile
// goes D-layout -> LDS -> A-layout (intra-wave, guarded by s_wait_dscnt).
// All register-fragment loops fully unrolled (avoid v_movrels indexing).
__global__ __launch_bounds__(256) void attn_flash(
    const __bf16* __restrict__ Q, const __bf16* __restrict__ Kb,
    const __bf16* __restrict__ Vb, __bf16* __restrict__ O) {
  __shared__ alignas(16) __bf16 vt[DH_][40];        // V^T tile: [d][s], stride 40 keeps 16B align
  __shared__ alignas(16) __bf16 plds[8][16 * 32];   // per-wave P tile (16 rows x 32 keys)

  const int wave = threadIdx.x >> 5;
  const int lane = threadIdx.x & 31;
  const int hf   = lane >> 4;
  const int l16  = lane & 15;
  const int bh = blockIdx.y;
  const int b = bh / H_, h = bh % H_, hk = h / REP_;
  const int qtile = blockIdx.x * 128 + wave * 16;

  // Q fragments: 16 rows x 128 d = 4 K-chunks of 32
  const __bf16* qrow = Q + ((size_t)(b * T_ + qtile + l16)) * (H_ * DH_) + h * DH_ + hf * 8;
  v16bf qf[4];
  #pragma unroll
  for (int f = 0; f < 4; ++f) qf[f] = load_frag(qrow + f * 32);

  v8f oacc[8];
  #pragma unroll
  for (int n = 0; n < 8; ++n) oacc[n] = (v8f){0.f,0.f,0.f,0.f,0.f,0.f,0.f,0.f};
  float mrow[8], lrow[8];
  #pragma unroll
  for (int j = 0; j < 8; ++j) { mrow[j] = -3.0e38f; lrow[j] = 0.f; }

  const float scale = 0.08838834764831845f;  // 1/sqrt(128)
  const int kend = blockIdx.x * 128 + 128;

  for (int ks = 0; ks < kend; ks += 32) {
    __syncthreads();
    // cooperative stage of V^T tile (32 keys x 128 dims)
    #pragma unroll
    for (int it = 0; it < 16; ++it) {
      int i = threadIdx.x + it * 256;
      int s = i >> 7, d = i & 127;
      vt[d][s] = Vb[((size_t)(b * T_ + ks + s)) * (HK_ * DH_) + hk * DH_ + d];
    }
    __syncthreads();

    if (ks < qtile + 16) {  // this key tile intersects the wave's causal range
      // S = Q K^T : two 16-key subtiles
      v8f s0 = {0.f,0.f,0.f,0.f,0.f,0.f,0.f,0.f};
      v8f s1 = {0.f,0.f,0.f,0.f,0.f,0.f,0.f,0.f};
      const __bf16* k0 = Kb + ((size_t)(b * T_ + ks + l16)) * (HK_ * DH_) + hk * DH_ + hf * 8;
      const __bf16* k1 = k0 + (size_t)16 * (HK_ * DH_);
      #pragma unroll
      for (int f = 0; f < 4; ++f) {
        v16bf kf0 = load_frag(k0 + f * 32);
        v16bf kf1 = load_frag(k1 + f * 32);
        s0 = __builtin_amdgcn_wmma_f32_16x16x32_bf16(false, qf[f], false, kf0, (short)0, s0, false, false);
        s1 = __builtin_amdgcn_wmma_f32_16x16x32_bf16(false, qf[f], false, kf1, (short)0, s1, false, false);
      }
      __bf16* pl = &plds[wave][0];
      #pragma unroll
      for (int j = 0; j < 8; ++j) {
        int t = qtile + hf * 8 + j;  // query row (D-layout: rows = vgpr + 8*(lane/16))
        float a0 = s0[j] * scale; if (ks + l16 > t)       a0 = -3.0e38f;
        float a1 = s1[j] * scale; if (ks + 16 + l16 > t)  a1 = -3.0e38f;
        // row-max across the 16 lanes holding this row's columns
        float rm = fmaxf(a0, a1);
        rm = fmaxf(rm, __shfl_xor(rm, 1, 32));
        rm = fmaxf(rm, __shfl_xor(rm, 2, 32));
        rm = fmaxf(rm, __shfl_xor(rm, 4, 32));
        rm = fmaxf(rm, __shfl_xor(rm, 8, 32));
        float newm  = fmaxf(mrow[j], rm);
        float alpha = __expf(mrow[j] - newm);
        mrow[j] = newm;
        float p0 = __expf(a0 - newm);
        float p1 = __expf(a1 - newm);
        float rs = p0 + p1;
        rs += __shfl_xor(rs, 1, 32);
        rs += __shfl_xor(rs, 2, 32);
        rs += __shfl_xor(rs, 4, 32);
        rs += __shfl_xor(rs, 8, 32);
        lrow[j] = lrow[j] * alpha + rs;
        #pragma unroll
        for (int n = 0; n < 8; ++n) oacc[n][j] *= alpha;
        // stash P row-major (16 x 32) for A-fragment reload
        pl[(hf * 8 + j) * 32 + l16]      = f2bf(p0);
        pl[(hf * 8 + j) * 32 + 16 + l16] = f2bf(p1);
      }
      asm volatile("s_wait_dscnt 0" ::: "memory");  // intra-wave LDS RAW fence
      v16bf pf = load_frag(pl + l16 * 32 + hf * 8);
      #pragma unroll
      for (int n = 0; n < 8; ++n) {
        v16bf vf = load_frag(&vt[n * 16 + l16][hf * 8]);
        oacc[n] = __builtin_amdgcn_wmma_f32_16x16x32_bf16(false, pf, false, vf, (short)0, oacc[n], false, false);
      }
    }
  }

  #pragma unroll
  for (int j = 0; j < 8; ++j) {
    float inv = 1.0f / lrow[j];
    size_t base = ((size_t)(b * T_ + qtile + hf * 8 + j)) * (H_ * DH_) + h * DH_;
    #pragma unroll
    for (int n = 0; n < 8; ++n)
      O[base + n * 16 + l16] = f2bf(oacc[n][j] * inv);
  }
}

// ----------------------------- SwiGLU combine ------------------------------
__global__ void silu_mul_kernel(const __bf16* __restrict__ g, const __bf16* __restrict__ u,
                                __bf16* __restrict__ out, long long n) {
  long long i = (long long)blockIdx.x * blockDim.x + threadIdx.x;
  long long stride = (long long)gridDim.x * blockDim.x;
  for (; i < n; i += stride) {
    float gv = bf2f(g[i]), uv = bf2f(u[i]);
    float s = gv / (1.f + __expf(-gv));
    out[i] = f2bf(s * uv);
  }
}

// ------------------------------- launcher ----------------------------------
extern "C" void kernel_launch(void* const* d_in, const int* in_sizes, int n_in,
                              void* d_out, int out_size, void* d_ws, size_t ws_size,
                              hipStream_t stream) {
  const float* x    = (const float*)d_in[0];
  const float* ln1  = (const float*)d_in[1];
  const float* ln2  = (const float*)d_in[2];
  const float* qUs  = (const float*)d_in[3];
  const float* qV   = (const float*)d_in[4];
  const float* kUs  = (const float*)d_in[5];
  const float* kV   = (const float*)d_in[6];
  const float* vUs  = (const float*)d_in[7];
  const float* vV   = (const float*)d_in[8];
  const float* oUs  = (const float*)d_in[9];
  const float* oV   = (const float*)d_in[10];
  const float* gUs  = (const float*)d_in[11];
  const float* gV   = (const float*)d_in[12];
  const float* uUs  = (const float*)d_in[13];
  const float* uV   = (const float*)d_in[14];
  const float* dUs  = (const float*)d_in[15];
  const float* dV   = (const float*)d_in[16];
  const float* rope = (const float*)d_in[17];
  float* out = (float*)d_out;

  char* p = (char*)d_ws;
  auto carve = [&](size_t bytes) { void* r = (void*)p; p += (bytes + 255) & ~(size_t)255; return r; };
  auto bfbuf = [&](size_t elems) { return (__bf16*)carve(elems * 2); };

  // bf16 weight copies
  __bf16* wqUs = bfbuf((size_t)H_ * DH_ * RQ_);
  __bf16* wqV  = bfbuf((size_t)H_ * RQ_ * D_);
  __bf16* wkUs = bfbuf((size_t)HK_ * DH_ * RKV_);
  __bf16* wkV  = bfbuf((size_t)HK_ * RKV_ * D_);
  __bf16* wvUs = bfbuf((size_t)HK_ * DH_ * RKV_);
  __bf16* wvV  = bfbuf((size_t)HK_ * RKV_ * D_);
  __bf16* woUs = bfbuf((size_t)D_ * RO_);
  __bf16* woV  = bfbuf((size_t)RO_ * D_);
  __bf16* wgUs = bfbuf((size_t)INTER_ * RFF_);
  __bf16* wgV  = bfbuf((size_t)RFF_ * D_);
  __bf16* wuUs = bfbuf((size_t)INTER_ * RFF_);
  __bf16* wuV  = bfbuf((size_t)RFF_ * D_);
  __bf16* wdUs = bfbuf((size_t)D_ * RFF_);
  __bf16* wdV  = bfbuf((size_t)RFF_ * INTER_);
  // activations
  __bf16* hbf   = bfbuf((size_t)MTOK_ * D_);
  __bf16* xrq   = bfbuf((size_t)MTOK_ * H_ * RQ_);
  __bf16* xrk   = bfbuf((size_t)MTOK_ * HK_ * RKV_);
  __bf16* xrv   = bfbuf((size_t)MTOK_ * HK_ * RKV_);
  __bf16* qbuf  = bfbuf((size_t)MTOK_ * H_ * DH_);
  __bf16* kbuf  = bfbuf((size_t)MTOK_ * HK_ * DH_);
  __bf16* vbuf  = bfbuf((size_t)MTOK_ * HK_ * DH_);
  __bf16* attnb = bfbuf((size_t)MTOK_ * D_);
  __bf16* orr   = bfbuf((size_t)MTOK_ * RO_);
  float*  h2    = (float*)carve((size_t)MTOK_ * D_ * 4);
  __bf16* mbf   = bfbuf((size_t)MTOK_ * D_);
  __bf16* grr   = bfbuf((size_t)MTOK_ * RFF_);
  __bf16* urr   = bfbuf((size_t)MTOK_ * RFF_);
  __bf16* gate  = bfbuf((size_t)MTOK_ * INTER_);
  __bf16* up    = bfbuf((size_t)MTOK_ * INTER_);
  __bf16* drr   = bfbuf((size_t)MTOK_ * RFF_);

  auto cast = [&](const float* src, __bf16* dst, long long n) {
    int grid = (int)(((n + 255) / 256) < 4096 ? ((n + 255) / 256) : 4096);
    cast_f32_bf16<<<grid, 256, 0, stream>>>(src, dst, n);
  };
  cast(qUs, wqUs, (long long)H_ * DH_ * RQ_);
  cast(qV,  wqV,  (long long)H_ * RQ_ * D_);
  cast(kUs, wkUs, (long long)HK_ * DH_ * RKV_);
  cast(kV,  wkV,  (long long)HK_ * RKV_ * D_);
  cast(vUs, wvUs, (long long)HK_ * DH_ * RKV_);
  cast(vV,  wvV,  (long long)HK_ * RKV_ * D_);
  cast(oUs, woUs, (long long)D_ * RO_);
  cast(oV,  woV,  (long long)RO_ * D_);
  cast(gUs, wgUs, (long long)INTER_ * RFF_);
  cast(gV,  wgV,  (long long)RFF_ * D_);
  cast(uUs, wuUs, (long long)INTER_ * RFF_);
  cast(uV,  wuV,  (long long)RFF_ * D_);
  cast(dUs, wdUs, (long long)D_ * RFF_);
  cast(dV,  wdV,  (long long)RFF_ * INTER_);

  // C(M,N) = A(M,K) @ W(N,K)^T, bf16 out. NT=4 when N%512==0 (A-frag reuse x4).
  auto gemm = [&](const __bf16* A, const __bf16* W, __bf16* C, int M, int N, int K,
                  int lda, int ldw, int ldc, long long sA, long long sW, long long sC, int batch) {
    if (N % 512 == 0) {
      dim3 g(N / 512, M / 16, batch);
      gemm_wmma_bf16<4, 0><<<g, 256, 0, stream>>>(A, W, (void*)C, (const float*)nullptr,
                                                  K, lda, ldw, ldc, sA, sW, sC, 0LL);
    } else {
      dim3 g(N / 128, M / 16, batch);
      gemm_wmma_bf16<1, 0><<<g, 256, 0, stream>>>(A, W, (void*)C, (const float*)nullptr,
                                                  K, lda, ldw, ldc, sA, sW, sC, 0LL);
    }
  };
  // fp32 out + residual (all users have N%512==0)
  auto gemm_res = [&](const __bf16* A, const __bf16* W, float* C, const float* R,
                      int M, int N, int K, int lda, int ldw, int ldc) {
    dim3 g(N / 512, M / 16, 1);
    gemm_wmma_bf16<4, 1><<<g, 256, 0, stream>>>(A, W, (void*)C, R, K, lda, ldw, ldc, 0LL, 0LL, 0LL, 0LL);
  };

  // 1) h = rmsnorm(x, ln1)
  rmsnorm_kernel<<<MTOK_, 256, 0, stream>>>(x, ln1, hbf, D_);

  // 2) low-rank projections: xr = h @ V^T (heads stacked), then per-head @ Us^T
  gemm(hbf, wqV, xrq, MTOK_, H_ * RQ_, D_, D_, D_, H_ * RQ_, 0, 0, 0, 1);
  gemm(hbf, wkV, xrk, MTOK_, HK_ * RKV_, D_, D_, D_, HK_ * RKV_, 0, 0, 0, 1);
  gemm(hbf, wvV, xrv, MTOK_, HK_ * RKV_, D_, D_, D_, HK_ * RKV_, 0, 0, 0, 1);
  gemm(xrq, wqUs, qbuf, MTOK_, DH_, RQ_, H_ * RQ_, RQ_, H_ * DH_, RQ_, (long long)DH_ * RQ_, DH_, H_);
  gemm(xrk, wkUs, kbuf, MTOK_, DH_, RKV_, HK_ * RKV_, RKV_, HK_ * DH_, RKV_, (long long)DH_ * RKV_, DH_, HK_);
  gemm(xrv, wvUs, vbuf, MTOK_, DH_, RKV_, HK_ * RKV_, RKV_, HK_ * DH_, RKV_, (long long)DH_ * RKV_, DH_, HK_);

  // 3) RoPE on q and k
  {
    long long tq = (long long)MTOK_ * H_ * 64;
    rope_kernel<<<(int)((tq + 255) / 256), 256, 0, stream>>>(qbuf, rope, H_, tq);
    long long tk = (long long)MTOK_ * HK_ * 64;
    rope_kernel<<<(int)((tk + 255) / 256), 256, 0, stream>>>(kbuf, rope, HK_, tk);
  }

  // 4) causal flash attention (GQA via hk = h/4 indexing)
  attn_flash<<<dim3(T_ / 128, B_ * H_), 256, 0, stream>>>(qbuf, kbuf, vbuf, attnb);

  // 5) output projection + residual -> h2 (fp32)
  gemm(attnb, woV, orr, MTOK_, RO_, D_, D_, D_, RO_, 0, 0, 0, 1);
  gemm_res(orr, woUs, h2, x, MTOK_, D_, RO_, RO_, RO_, D_);

  // 6) rmsnorm2 -> m
  rmsnorm_kernel<<<MTOK_, 256, 0, stream>>>(h2, ln2, mbf, D_);

  // 7) SwiGLU MLP (low-rank both stages)
  gemm(mbf, wgV, grr, MTOK_, RFF_, D_, D_, D_, RFF_, 0, 0, 0, 1);
  gemm(mbf, wuV, urr, MTOK_, RFF_, D_, D_, D_, RFF_, 0, 0, 0, 1);
  gemm(grr, wgUs, gate, MTOK_, INTER_, RFF_, RFF_, RFF_, INTER_, 0, 0, 0, 1);
  gemm(urr, wuUs, up,   MTOK_, INTER_, RFF_, RFF_, RFF_, INTER_, 0, 0, 0, 1);
  {
    long long n = (long long)MTOK_ * INTER_;
    silu_mul_kernel<<<8192, 256, 0, stream>>>(gate, up, gate, n);  // in-place: gate <- silu(gate)*up
  }
  gemm(gate, wdV, drr, MTOK_, RFF_, INTER_, INTER_, INTER_, RFF_, 0, 0, 0, 1);

  // 8) final down-proj + residual -> out (fp32)
  gemm_res(drr, wdUs, out, h2, MTOK_, D_, RFF_, RFF_, RFF_, D_);
}
